// MultiHeadAttention_4380866641966
// MI455X (gfx1250) — compile-verified, matrix operand order checked by reading, not run on hardware
//
#include <hip/hip_runtime.h>
#include <hip/hip_bf16.h>

typedef __attribute__((ext_vector_type(16))) _Float16 v16h;
typedef __attribute__((ext_vector_type(8)))  float    v8f;
typedef __attribute__((ext_vector_type(4)))  int      v4i;

#define B_  8
#define N_  1024
#define E_  768
#define H_  8
#define D_  96
#define M_TOT (B_ * N_)            // 8192 rows
#define QKV_COLS (3 * E_)          // 2304

// ---------------------------------------------------------------------------
// CDNA5 async global->LDS copy (ASYNCcnt path), with safe fallback.
// Builtin signature (probe-confirmed by round-2 diagnostic):
//   void __builtin_amdgcn_global_load_async_to_lds_b128(
//       v4i addrspace(1)* gsrc, v4i addrspace(3)* lds, imm int off, imm int cpol)
// ---------------------------------------------------------------------------
typedef __attribute__((address_space(1))) v4i as1_v4i;
typedef __attribute__((address_space(3))) v4i as3_v4i;

#if __has_builtin(__builtin_amdgcn_global_load_async_to_lds_b128)
#define ASYNC_LDS 1
__device__ __forceinline__ void async_b128(const void* gsrc, void* ldst) {
  __builtin_amdgcn_global_load_async_to_lds_b128(
      (as1_v4i*)gsrc, (as3_v4i*)ldst, /*offset=*/0, /*cpol=*/0);
}
#else
#define ASYNC_LDS 0
__device__ __forceinline__ void async_b128(const void* gsrc, void* ldst) {
  *(uint4*)ldst = *(const uint4*)gsrc;   // synchronous fallback
}
#endif

__device__ __forceinline__ void wait_async0() {
#if __has_builtin(__builtin_amdgcn_s_wait_asynccnt)
  __builtin_amdgcn_s_wait_asynccnt(0);
#elif ASYNC_LDS
  asm volatile("s_wait_asynccnt 0" ::: "memory");
#endif
}

// ---------------------------------------------------------------------------
// WMMA helper: D = A(16x32 f16) * B(32x16 f16) + C(16x16 f32)
// ---------------------------------------------------------------------------
__device__ __forceinline__ v8f wmma_f16(v16h a, v16h b, v8f c) {
  return __builtin_amdgcn_wmma_f32_16x16x32_f16(
      /*neg_a=*/false, a, /*neg_b=*/false, b,
      /*c_mod=*/(short)0, c, /*reuse_a=*/false, /*reuse_b=*/false);
}

// Load a 16x32 f16 A-fragment (or mirrored B-fragment) from LDS.
// Lanes 0-15 -> row = row0+lane, halves = K0..7,K16..23
// Lanes 16-31 -> row = row0+lane-16, halves = K8..15,K24..31
__device__ __forceinline__ v16h lds_frag(const _Float16* base, int row0,
                                         int rowStrideHalves, int kBaseHalves,
                                         int lane) {
  int r = row0 + (lane & 15);
  int k = kBaseHalves + ((lane >> 4) << 3);
  const _Float16* p = base + r * rowStrideHalves + k;
  union { uint4 q[2]; v16h h; } u;
  u.q[0] = *reinterpret_cast<const uint4*>(p);
  u.q[1] = *reinterpret_cast<const uint4*>(p + 16);
  return u.h;
}

// ---------------------------------------------------------------------------
// Kernel 0: pack / convert / de-interleave weights
// ---------------------------------------------------------------------------
__global__ void mha_pack(const float* __restrict__ x,
                         const float* __restrict__ Wqkv,
                         const float* __restrict__ bqkv,
                         const float* __restrict__ Wproj,
                         _Float16* __restrict__ Xh,
                         _Float16* __restrict__ Wt,
                         float* __restrict__ bcat,
                         _Float16* __restrict__ Wpt) {
  const long stride = (long)gridDim.x * blockDim.x;
  const long i0 = (long)blockIdx.x * blockDim.x + threadIdx.x;

  for (long t = i0; t < (long)M_TOT * E_; t += stride)
    Xh[t] = (_Float16)x[t];

  for (long t = i0; t < (long)QKV_COLS * E_; t += stride) {
    int rc = (int)(t / E_), k = (int)(t % E_);
    int which = rc / E_, e = rc % E_;
    int h = e / D_, d = e % D_;
    int col = h * (3 * D_) + d * 3 + which;   // (h d qkv) fastest-varying
    Wt[t] = (_Float16)Wqkv[(long)k * QKV_COLS + col];
  }

  for (long t = i0; t < QKV_COLS; t += stride) {
    int which = (int)(t / E_), e = (int)(t % E_);
    int h = e / D_, d = e % D_;
    bcat[t] = bqkv[h * (3 * D_) + d * 3 + which];
  }

  for (long t = i0; t < (long)E_ * E_; t += stride) {
    int oc = (int)(t / E_), k = (int)(t % E_);
    Wpt[t] = (_Float16)Wproj[(long)k * E_ + oc];
  }
}

// ---------------------------------------------------------------------------
// Kernel 1: QKV GEMM  C[8192][2304] = Xh @ Wt^T + bcat, scattered to Q/K/V.
// 256 threads = 8 waves, tile 128x128, K-step 32, DOUBLE-BUFFERED async LDS.
// Q is pre-scaled by E^-0.5 here so flash softmax skips the multiply.
// ---------------------------------------------------------------------------
__global__ void mha_qkv_gemm(const _Float16* __restrict__ Xh,
                             const _Float16* __restrict__ Wt,
                             const float* __restrict__ bcat,
                             _Float16* __restrict__ Qh,
                             _Float16* __restrict__ Kh,
                             _Float16* __restrict__ Vh) {
  __shared__ _Float16 As[2][128 * 32];
  __shared__ _Float16 Bs[2][128 * 32];
  const int tid = threadIdx.x, lane = tid & 31, wave = tid >> 5;
  const int wm = wave >> 1, wn = wave & 1;
  const int m0 = blockIdx.x * 128;
  const int c0 = blockIdx.y * 128;

  auto issue_tile = [&](int buf, int k0) {
    for (int i = tid; i < 512; i += 256) {       // 128 rows x 4 uint4
      int r = i >> 2, cb = (i & 3) << 3;
      async_b128(&Xh[(long)(m0 + r) * E_ + k0 + cb], &As[buf][r * 32 + cb]);
      async_b128(&Wt[(long)(c0 + r) * E_ + k0 + cb], &Bs[buf][r * 32 + cb]);
    }
  };

  v8f acc[2][4] = {};
  issue_tile(0, 0);

  const int NITER = E_ / 32;                      // 24
  for (int it = 0; it < NITER; ++it) {
    const int cur = it & 1;
    wait_async0();                                // tile `it` resident
    __syncthreads();                              // everyone done with buf cur^1
    if (it + 1 < NITER) issue_tile(cur ^ 1, (it + 1) * 32);  // overlap with WMMA

    v16h af0 = lds_frag(As[cur], wm * 32 + 0,  32, 0, lane);
    v16h af1 = lds_frag(As[cur], wm * 32 + 16, 32, 0, lane);
#pragma unroll
    for (int nt = 0; nt < 4; ++nt) {
      v16h bf = lds_frag(Bs[cur], wn * 64 + nt * 16, 32, 0, lane);
      acc[0][nt] = wmma_f16(af0, bf, acc[0][nt]);
      acc[1][nt] = wmma_f16(af1, bf, acc[1][nt]);
    }
  }

  const float scale = 0.03608439182435161f;       // E^-0.5, folded into Q
#pragma unroll
  for (int mi = 0; mi < 2; ++mi)
#pragma unroll
    for (int nt = 0; nt < 4; ++nt)
#pragma unroll
      for (int vi = 0; vi < 8; ++vi) {
        int row = m0 + wm * 32 + mi * 16 + vi + ((lane >> 4) << 3);
        int c   = c0 + wn * 64 + nt * 16 + (lane & 15);
        float val = acc[mi][nt][vi] + bcat[c];
        int which = c / E_, e = c % E_;
        int b = row >> 10, n = row & (N_ - 1);
        int h = e / D_, d = e % D_;
        long idx = (((long)(b * H_ + h) * N_) + n) * D_ + d;
        if (which == 0)       Qh[idx] = (_Float16)(val * scale);
        else if (which == 1)  Kh[idx] = (_Float16)val;
        else                  Vh[idx] = (_Float16)val;
      }
}

// ---------------------------------------------------------------------------
// Kernel 2: flash attention. grid = (N/128, B*H), block = 256 (8 waves).
// Each wave owns 16 query rows -> per-row softmax stats stay in one wave half.
// ---------------------------------------------------------------------------
__global__ void mha_flash(const _Float16* __restrict__ Qh,
                          const _Float16* __restrict__ Kh,
                          const _Float16* __restrict__ Vh,
                          _Float16* __restrict__ Oh) {
  extern __shared__ _Float16 smem[];
  _Float16* Qs = smem;                 // [128][96]
  _Float16* Ks = Qs + 128 * 96;        // [128][96]   (keys row-major = B^T)
  _Float16* Vt = Ks + 128 * 96;        // [96][128]   (V transposed: [d][key])
  _Float16* Ps = Vt + 96 * 128;        // 8 waves x [16][128]

  const int tid = threadIdx.x, lane = tid & 31, wave = tid >> 5;
  const int bh = blockIdx.y;           // b*H + h
  const int q0 = blockIdx.x * 128;

  // Async-load Q tile (contiguous 128*96 span); Q already carries the scale.
  const _Float16* Qg = Qh + ((long)bh * N_ + q0) * D_;
  for (int i = tid; i < 1536; i += 256)
    async_b128(&Qg[i * 8], &Qs[i * 8]);

  float m_i[8], l_i[8];
  v8f o[6] = {};
#pragma unroll
  for (int vi = 0; vi < 8; ++vi) { m_i[vi] = -3.0e38f; l_i[vi] = 0.f; }

  for (int kt = 0; kt < N_ / 128; ++kt) {
    const _Float16* Kg = Kh + ((long)bh * N_ + kt * 128) * D_;
    const _Float16* Vg = Vh + ((long)bh * N_ + kt * 128) * D_;
    __syncthreads();                              // done reading Ks/Vt/Ps
    for (int i = tid; i < 1536; i += 256)
      async_b128(&Kg[i * 8], &Ks[i * 8]);
    for (int i = tid; i < 1536; i += 256) {       // V transpose scatter
      int key = i / 12, d0 = (i % 12) * 8;        // 12 uint4 per 96-wide row
      union { uint4 q; _Float16 hh[8]; } u;
      u.q = *(const uint4*)&Vg[key * D_ + d0];
#pragma unroll
      for (int j = 0; j < 8; ++j) Vt[(d0 + j) * 128 + key] = u.hh[j];
    }
    wait_async0();
    __syncthreads();

    // S = Q * K^T for this wave's 16 rows x 128 keys
    v8f s[8] = {};
    v16h aq[3];
#pragma unroll
    for (int kk = 0; kk < 3; ++kk) aq[kk] = lds_frag(Qs, wave * 16, 96, kk * 32, lane);
#pragma unroll
    for (int nt = 0; nt < 8; ++nt)
#pragma unroll
      for (int kk = 0; kk < 3; ++kk) {
        v16h bf = lds_frag(Ks, nt * 16, 96, kk * 32, lane);
        s[nt] = wmma_f16(aq[kk], bf, s[nt]);
      }

    // Online softmax per row (row = vi + 8*(lane>=16); 16 lanes share a row)
#pragma unroll
    for (int vi = 0; vi < 8; ++vi) {
      float mblk = -3.0e38f;
#pragma unroll
      for (int nt = 0; nt < 8; ++nt) mblk = fmaxf(mblk, s[nt][vi]);
#pragma unroll
      for (int off = 8; off >= 1; off >>= 1)
        mblk = fmaxf(mblk, __shfl_xor(mblk, off, 32));
      float mnew  = fmaxf(m_i[vi], mblk);
      float alpha = __expf(m_i[vi] - mnew);
      float lsum  = 0.f;
      int row = vi + ((lane >> 4) << 3);
#pragma unroll
      for (int nt = 0; nt < 8; ++nt) {
        float p = __expf(s[nt][vi] - mnew);
        lsum += p;
        Ps[wave * 2048 + row * 128 + nt * 16 + (lane & 15)] = (_Float16)p;
      }
#pragma unroll
      for (int off = 8; off >= 1; off >>= 1) lsum += __shfl_xor(lsum, off, 32);
      l_i[vi] = l_i[vi] * alpha + lsum;
      m_i[vi] = mnew;
#pragma unroll
      for (int nt = 0; nt < 6; ++nt) o[nt][vi] *= alpha;
    }
    __syncthreads();   // Ps (C-layout -> A-layout transpose staging) visible

    // O += P * V   (K-dim = 128 keys = 4 WMMA steps)
#pragma unroll
    for (int kk = 0; kk < 4; ++kk) {
      v16h ap = lds_frag(Ps + wave * 2048, 0, 128, kk * 32, lane);
#pragma unroll
      for (int nt = 0; nt < 6; ++nt) {
        v16h bf = lds_frag(Vt, nt * 16, 128, kk * 32, lane);
        o[nt] = wmma_f16(ap, bf, o[nt]);
      }
    }
  }

  // Epilogue: normalize, write [B,N,E] f16 for the projection GEMM
  const int b = bh >> 3, h = bh & 7;
#pragma unroll
  for (int vi = 0; vi < 8; ++vi) {
    float inv = 1.0f / l_i[vi];
    int row = q0 + wave * 16 + vi + ((lane >> 4) << 3);
#pragma unroll
    for (int nt = 0; nt < 6; ++nt) {
      int col = h * D_ + nt * 16 + (lane & 15);
      Oh[((long)b * N_ + row) * E_ + col] = (_Float16)(o[nt][vi] * inv);
    }
  }
}

// ---------------------------------------------------------------------------
// Kernel 3: projection GEMM  out[8192][768] = Attn @ Wproj + bproj (f32 out)
// Same double-buffered async-LDS pipeline as kernel 1.
// ---------------------------------------------------------------------------
__global__ void mha_proj_gemm(const _Float16* __restrict__ Ah,
                              const _Float16* __restrict__ Wpt,
                              const float* __restrict__ bias,
                              float* __restrict__ Out) {
  __shared__ _Float16 As[2][128 * 32];
  __shared__ _Float16 Bs[2][128 * 32];
  const int tid = threadIdx.x, lane = tid & 31, wave = tid >> 5;
  const int wm = wave >> 1, wn = wave & 1;
  const int m0 = blockIdx.x * 128;
  const int c0 = blockIdx.y * 128;

  auto issue_tile = [&](int buf, int k0) {
    for (int i = tid; i < 512; i += 256) {
      int r = i >> 2, cb = (i & 3) << 3;
      async_b128(&Ah[(long)(m0 + r) * E_ + k0 + cb],  &As[buf][r * 32 + cb]);
      async_b128(&Wpt[(long)(c0 + r) * E_ + k0 + cb], &Bs[buf][r * 32 + cb]);
    }
  };

  v8f acc[2][4] = {};
  issue_tile(0, 0);

  const int NITER = E_ / 32;
  for (int it = 0; it < NITER; ++it) {
    const int cur = it & 1;
    wait_async0();
    __syncthreads();
    if (it + 1 < NITER) issue_tile(cur ^ 1, (it + 1) * 32);

    v16h af0 = lds_frag(As[cur], wm * 32 + 0,  32, 0, lane);
    v16h af1 = lds_frag(As[cur], wm * 32 + 16, 32, 0, lane);
#pragma unroll
    for (int nt = 0; nt < 4; ++nt) {
      v16h bf = lds_frag(Bs[cur], wn * 64 + nt * 16, 32, 0, lane);
      acc[0][nt] = wmma_f16(af0, bf, acc[0][nt]);
      acc[1][nt] = wmma_f16(af1, bf, acc[1][nt]);
    }
  }

#pragma unroll
  for (int mi = 0; mi < 2; ++mi)
#pragma unroll
    for (int nt = 0; nt < 4; ++nt)
#pragma unroll
      for (int vi = 0; vi < 8; ++vi) {
        int row = m0 + wm * 32 + mi * 16 + vi + ((lane >> 4) << 3);
        int c   = c0 + wn * 64 + nt * 16 + (lane & 15);
        Out[(long)row * E_ + c] = acc[mi][nt][vi] + bias[c];
      }
}

// ---------------------------------------------------------------------------
// Host launcher
// ---------------------------------------------------------------------------
extern "C" void kernel_launch(void* const* d_in, const int* in_sizes, int n_in,
                              void* d_out, int out_size, void* d_ws, size_t ws_size,
                              hipStream_t stream) {
  const float* x     = (const float*)d_in[0];
  const float* Wqkv  = (const float*)d_in[1];
  const float* bqkv  = (const float*)d_in[2];
  const float* Wproj = (const float*)d_in[3];
  const float* bproj = (const float*)d_in[4];
  float* out = (float*)d_out;

  char* ws = (char*)d_ws;
  size_t off = 0;
  auto alloc = [&](size_t bytes) {
    char* p = ws + off;
    off = (off + bytes + 255) & ~(size_t)255;
    return p;
  };
  _Float16* Xh   = (_Float16*)alloc((size_t)M_TOT * E_ * 2);
  _Float16* Wt   = (_Float16*)alloc((size_t)QKV_COLS * E_ * 2);
  float*    bcat = (float*)   alloc((size_t)QKV_COLS * 4);
  _Float16* Wpt  = (_Float16*)alloc((size_t)E_ * E_ * 2);
  _Float16* Qh   = (_Float16*)alloc((size_t)B_ * H_ * N_ * D_ * 2);
  _Float16* Kh   = (_Float16*)alloc((size_t)B_ * H_ * N_ * D_ * 2);
  _Float16* Vh   = (_Float16*)alloc((size_t)B_ * H_ * N_ * D_ * 2);
  _Float16* Ah   = (_Float16*)alloc((size_t)M_TOT * E_ * 2);

  mha_pack<<<4096, 256, 0, stream>>>(x, Wqkv, bqkv, Wproj, Xh, Wt, bcat, Wpt);

  mha_qkv_gemm<<<dim3(M_TOT / 128, QKV_COLS / 128), 256, 0, stream>>>(
      Xh, Wt, bcat, Qh, Kh, Vh);

  const size_t smem = (size_t)(128 * 96 + 128 * 96 + 96 * 128 + 8 * 16 * 128) * 2; // 104 KB
  mha_flash<<<dim3(N_ / 128, B_ * H_), 256, smem, stream>>>(Qh, Kh, Vh, Ah);

  mha_proj_gemm<<<dim3(M_TOT / 128, E_ / 128), 256, 0, stream>>>(Ah, Wpt, bproj, out);
}
